// Contrastive_loss_1632087572636
// MI455X (gfx1250) — matrix operand order repaired and checked by reading, not compiled, not run on hardware
//
#include <hip/hip_runtime.h>
#include <hip/hip_bf16.h>

// ---------------- problem constants ----------------
#define BATCH   4096
#define NROW    8192        // 2*BATCH
#define DDIM    256
#define INV_T   2.0f        // 1/TEMPERATURE
#define NCHUNK  8
#define COLS_PER_CHUNK (NROW / NCHUNK)   // 1024
#define NTILES  (COLS_PER_CHUNK / 16)    // 64 (even -> clean unroll-by-2)
#define ROWS_PER_WG    128               // 8 waves * 16 rows
#define KSTEPS  (DDIM / 32)              // 8

typedef __attribute__((ext_vector_type(16))) __bf16 v16bf;
typedef __attribute__((ext_vector_type(8)))  float  v8f;

struct __align__(16) U4 { unsigned int x, y, z, w; };

union Frag {
    v16bf v;
    U4    q[2];
};

// f32 -> bf16 round-to-nearest-even (inputs are finite gaussians, no NaN path needed)
__device__ __forceinline__ unsigned short f2bf(float f) {
    unsigned int u = __float_as_uint(f);
    u += 0x7FFFu + ((u >> 16) & 1u);
    return (unsigned short)(u >> 16);
}

// unpack a dword holding two bf16 into two floats
__device__ __forceinline__ void bf2x(unsigned int w, float& lo, float& hi) {
    lo = __uint_as_float(w << 16);
    hi = __uint_as_float(w & 0xFFFF0000u);
}

// ---------------- phase 1: fp32 row norm -> scaled bf16 zhat ----------------
// zhat[row] = z[row] * sqrt(1/T) / ||z[row]||  =>  WMMA dot == sim in [-2, 2].
__global__ __launch_bounds__(256)
void nt_xent_normalize(const float* __restrict__ zi,
                       const float* __restrict__ zj,
                       unsigned short* __restrict__ zhat) {
    const int lane = threadIdx.x & 31;
    const int wid  = threadIdx.x >> 5;
    const int row  = blockIdx.x * 8 + wid;           // 0..NROW-1

    const float* src = (row < BATCH) ? (zi + (size_t)row * DDIM)
                                     : (zj + (size_t)(row - BATCH) * DDIM);
    float4 f0 = ((const float4*)src)[lane * 2 + 0];
    float4 f1 = ((const float4*)src)[lane * 2 + 1];

    float ss = f0.x*f0.x + f0.y*f0.y + f0.z*f0.z + f0.w*f0.w
             + f1.x*f1.x + f1.y*f1.y + f1.z*f1.z + f1.w*f1.w;
    #pragma unroll
    for (int m = 16; m >= 1; m >>= 1) ss += __shfl_xor(ss, m, 32);

    const float scale = sqrtf(INV_T) * rsqrtf(fmaxf(ss, 1e-16f));

    U4 o;
    o.x = (unsigned int)f2bf(f0.x * scale) | ((unsigned int)f2bf(f0.y * scale) << 16);
    o.y = (unsigned int)f2bf(f0.z * scale) | ((unsigned int)f2bf(f0.w * scale) << 16);
    o.z = (unsigned int)f2bf(f1.x * scale) | ((unsigned int)f2bf(f1.y * scale) << 16);
    o.w = (unsigned int)f2bf(f1.z * scale) | ((unsigned int)f2bf(f1.w * scale) << 16);
    ((U4*)(zhat + (size_t)row * DDIM))[lane] = o;
}

// ---------------- phase 1b: diag & positive dot products ----------------
// diag[i] = zhat_i . zhat_i , pos[i] = zhat_i . zhat_{(i+B)%N}, fp32 accumulate.
// Matches the WMMA-produced entries to within accumulation-order rounding.
__global__ __launch_bounds__(256)
void nt_xent_pairdot(const unsigned short* __restrict__ zhat,
                     float* __restrict__ diag,
                     float* __restrict__ pos) {
    const int lane = threadIdx.x & 31;
    const int wid  = threadIdx.x >> 5;
    const int row  = blockIdx.x * 8 + wid;
    const int prow = (row + BATCH) & (NROW - 1);

    U4 av = ((const U4*)(zhat + (size_t)row  * DDIM))[lane];
    U4 bv = ((const U4*)(zhat + (size_t)prow * DDIM))[lane];

    float dd = 0.0f, pp = 0.0f;
    const unsigned int aw[4] = {av.x, av.y, av.z, av.w};
    const unsigned int bw[4] = {bv.x, bv.y, bv.z, bv.w};
    #pragma unroll
    for (int t = 0; t < 4; ++t) {
        float al, ah, bl, bh;
        bf2x(aw[t], al, ah);
        bf2x(bw[t], bl, bh);
        dd += al * al + ah * ah;
        pp += al * bl + ah * bh;
    }
    #pragma unroll
    for (int m = 16; m >= 1; m >>= 1) {
        dd += __shfl_xor(dd, m, 32);
        pp += __shfl_xor(pp, m, 32);
    }
    if (lane == 0) {
        diag[row] = dd;
        pos[row]  = pp;
    }
}

// ---------------- phase 2: WMMA Gram tiles + streaming exp-sum ----------------
// grid = (NROW/128, NCHUNK); 256 threads = 8 waves, 16 rows per wave.
// A (16 rows x K=256) resident in 64 VGPRs; B double-buffered across column
// tiles so tile ct+1's 16 b128 loads issue before tile ct's 8 WMMAs.
// Per ISA 7.12.2 (16-bit, wave32):
//   A 16x32 : lane m=lane&15 (M=m), hi=lane>>4; VGPR0-3: K=hi*8+{0..7},
//             VGPR4-7: K=16+hi*8+{0..7}.
//   B 32x16 : lane m holds column N=m, same K order as A (hi*8+{0..7}, 16+hi*8+{0..7}).
//   C/D f32 : acc[r] = S[row_base + r + 8*hi][col_base + m].
__global__ __launch_bounds__(256)
void nt_xent_gram(const unsigned short* __restrict__ zhat,
                  float* __restrict__ sum_part) {
    const int lane = threadIdx.x & 31;
    const int wid  = threadIdx.x >> 5;
    const int m    = lane & 15;
    const int hi   = lane >> 4;
    const int row_base = blockIdx.x * ROWS_PER_WG + wid * 16;
    const int col_chunk_base = blockIdx.y * COLS_PER_CHUNK;

    // resident A fragments over full K
    Frag a[KSTEPS];
    {
        const unsigned short* arow = zhat + (size_t)(row_base + m) * DDIM;
        #pragma unroll
        for (int ks = 0; ks < KSTEPS; ++ks) {
            const unsigned short* p = arow + ks * 32 + hi * 8;
            a[ks].q[0] = *(const U4*)(p);
            a[ks].q[1] = *(const U4*)(p + 16);
        }
    }

    const unsigned short* bbase =
        zhat + (size_t)(col_chunk_base + m) * DDIM + hi * 8;

    auto loadB = [&](Frag* dst, int ct) {
        const unsigned short* bcol = bbase + (size_t)ct * 16 * DDIM;
        __builtin_prefetch(bcol + 32 * DDIM, 0, 1);   // 2 tiles ahead
        #pragma unroll
        for (int ks = 0; ks < KSTEPS; ++ks) {
            dst[ks].q[0] = *(const U4*)(bcol + ks * 32);
            dst[ks].q[1] = *(const U4*)(bcol + ks * 32 + 16);
        }
    };

    float se[8];
    #pragma unroll
    for (int r = 0; r < 8; ++r) se[r] = 0.0f;

    auto computeTile = [&](const Frag* b) {
        v8f acc = {};
        #pragma unroll
        for (int ks = 0; ks < KSTEPS; ++ks) {
            acc = __builtin_amdgcn_wmma_f32_16x16x32_bf16(
                      false, a[ks].v, false, b[ks].v, (short)0, acc, false, false);
        }
        #pragma unroll
        for (int r = 0; r < 8; ++r) se[r] += __expf(acc[r]);
    };

    Frag b0[KSTEPS], b1[KSTEPS];
    loadB(b0, 0);
    for (int ct = 0; ct < NTILES; ct += 2) {
        loadB(b1, ct + 1);          // issue next tile's loads before computing
        computeTile(b0);            // tile ct
        if (ct + 2 < NTILES) loadB(b0, ct + 2);
        computeTile(b1);            // tile ct+1
    }

    // reduce exp-sums across the 16 lanes of each half-wave
    #pragma unroll
    for (int r = 0; r < 8; ++r) {
        #pragma unroll
        for (int msk = 1; msk <= 8; msk <<= 1)
            se[r] += __shfl_xor(se[r], msk, 32);
    }
    if (m == 0) {   // lanes 0 / 16 hold rows r / r+8
        #pragma unroll
        for (int r = 0; r < 8; ++r) {
            const int g = row_base + r + 8 * hi;
            sum_part[(size_t)blockIdx.y * NROW + g] = se[r];
        }
    }
}

// ---------------- phase 3: deterministic final reduction ----------------
__global__ __launch_bounds__(256)
void nt_xent_finalize(const float* __restrict__ sum_part,
                      const float* __restrict__ diag,
                      const float* __restrict__ pos,
                      float* __restrict__ out) {
    __shared__ float red[256];
    const int tid = threadIdx.x;
    float acc = 0.0f;
    for (int i = tid; i < NROW; i += 256) {
        float s = 0.0f;
        #pragma unroll
        for (int c = 0; c < NCHUNK; ++c) s += sum_part[(size_t)c * NROW + i];
        // remove self-term from denominator; sim bounded in [-2,2] -> no max trick
        acc += logf(s - __expf(diag[i])) - pos[i];
    }
    red[tid] = acc;
    __syncthreads();
    for (int st = 128; st > 0; st >>= 1) {
        if (tid < st) red[tid] += red[tid + st];
        __syncthreads();
    }
    if (tid == 0) out[0] = red[0] / (float)NROW;
}

// ---------------- host entry ----------------
extern "C" void kernel_launch(void* const* d_in, const int* in_sizes, int n_in,
                              void* d_out, int out_size, void* d_ws, size_t ws_size,
                              hipStream_t stream) {
    const float* zi = (const float*)d_in[0];
    const float* zj = (const float*)d_in[1];
    float* out = (float*)d_out;

    // ws layout: [zhat bf16: 4MB][sum_part: NCHUNK*NROW f32][diag: NROW f32][pos: NROW f32]
    unsigned short* zhat = (unsigned short*)d_ws;
    float* sum_part = (float*)((char*)d_ws + (size_t)NROW * DDIM * sizeof(unsigned short));
    float* diag     = sum_part + (size_t)NCHUNK * NROW;
    float* pos      = diag + NROW;

    nt_xent_normalize<<<dim3(NROW / 8), dim3(256), 0, stream>>>(zi, zj, zhat);
    nt_xent_pairdot  <<<dim3(NROW / 8), dim3(256), 0, stream>>>(zhat, diag, pos);
    nt_xent_gram     <<<dim3(NROW / ROWS_PER_WG, NCHUNK), dim3(256), 0, stream>>>(zhat, sum_part);
    nt_xent_finalize <<<dim3(1), dim3(256), 0, stream>>>(sum_part, diag, pos, out);
}
